// Conv2dKan_81432579932372
// MI455X (gfx1250) — matrix operand order repaired and checked by reading, not compiled, not run on hardware
//
#include <hip/hip_runtime.h>
#include <cstdint>

typedef __attribute__((ext_vector_type(16))) _Float16 v16h;
typedef __attribute__((ext_vector_type(8)))  _Float16 v8h;
typedef __attribute__((ext_vector_type(8)))  float    v8f;

#define CIN    64
#define COUT   64
#define KK     9
#define BASIS  8
#define NF     10          // feature slots per (i,k): silu, T0..T7, one zero pad
#define HH     32
#define WW     32
#define NTILE  64          // output columns (n = b*1024 + y*32 + x) per workgroup
#define FS_STRIDE 104      // halfs per n-row in LDS (52 dwords -> conflict-free)
#define WF_THREADS (CIN*3*4*32)     // one thread per 16-half A-fragment lane

#define DEV_INLINE __device__ __forceinline__

// ---------------------------------------------------------------------------
// Prep: fuse w and c into f16 weights, pre-swizzled into the exact 16x32 f16
// A-fragment layout (lanes 0-15: M=lane, K = s + (s<8?0:8); lanes 16-31: +8).
// Reduction index per channel: rp = k*10 + j (j=0 silu, j=1..8 Cheb, j=9 pad),
// rp 90..95 pad -> zero.  Layout: [i][kc][mt][lane][16 halfs].
// ---------------------------------------------------------------------------
__global__ void kan_prep_weights(const float* __restrict__ w,
                                 const float* __restrict__ c,
                                 _Float16* __restrict__ wf) {
  int t = blockIdx.x * blockDim.x + threadIdx.x;
  if (t >= WF_THREADS) return;
  int lane = t & 31;
  int mt   = (t >> 5) & 3;
  int rest = t >> 7;            // i*3 + kc
  int kc   = rest % 3;
  int i    = rest / 3;
  int o    = mt * 16 + (lane & 15);
  int khi  = (lane < 16) ? 0 : 8;

  union { v16h v; _Float16 h[16]; } vals;
  #pragma unroll
  for (int s = 0; s < 16; ++s) {
    int kl = s + ((s < 8) ? 0 : 8) + khi;   // 0..31
    int rp = kc * 32 + kl;                  // 0..95
    float val = 0.0f;
    if (rp < 90) {
      int k = rp / NF;
      int j = rp % NF;
      if (j < 9) {
        float wv = w[(i * COUT + o) * KK + k];
        val = (j == 0) ? wv
                       : wv * c[((i * COUT + o) * KK + k) * BASIS + (j - 1)];
      }
    }
    vals.h[s] = (_Float16)val;
  }
  *(v16h*)(wf + (size_t)t * 16) = vals.v;
}

// ---------------------------------------------------------------------------
// Feature evaluation: silu(p) and Chebyshev T0..T7 of tanh(p), written as
// five packed dwords into one LDS row.  One v_exp_f32 + two v_rcp_f32:
//   e   = exp(-p); sigmoid = 1/(1+e); tanh = (1-e^2)/(1+e^2)
// ---------------------------------------------------------------------------
DEV_INLINE void write_features(_Float16* row, float p) {
  float e1 = __expf(-p);
  float sg = __builtin_amdgcn_rcpf(1.0f + e1);
  float f0 = p * sg;                       // silu
  float e2 = e1 * e1;                      // exp(-2p)
  float t  = (1.0f - e2) * __builtin_amdgcn_rcpf(1.0f + e2);
  float t2 = 2.0f * t;
  float T0 = 1.0f, T1 = t;
  float T2 = t2 * T1 - T0;
  float T3 = t2 * T2 - T1;
  float T4 = t2 * T3 - T2;
  float T5 = t2 * T4 - T3;
  float T6 = t2 * T5 - T4;
  float T7 = t2 * T6 - T5;
  union { _Float16 h[2]; uint32_t u; } pk;
  pk.h[0] = (_Float16)f0; pk.h[1] = (_Float16)T0; *(uint32_t*)&row[0] = pk.u;
  pk.h[0] = (_Float16)T1; pk.h[1] = (_Float16)T2; *(uint32_t*)&row[2] = pk.u;
  pk.h[0] = (_Float16)T3; pk.h[1] = (_Float16)T4; *(uint32_t*)&row[4] = pk.u;
  pk.h[0] = (_Float16)T5; pk.h[1] = (_Float16)T6; *(uint32_t*)&row[6] = pk.u;
  pk.h[0] = (_Float16)T7; pk.h[1] = (_Float16)0.0f; *(uint32_t*)&row[8] = pk.u;
}

// One unfolded-patch point: pi in [0,576) -> (k = dy*3+dx, n = yl*32+xc)
DEV_INLINE void do_point(const float* __restrict__ xch, int y0, int pi,
                         _Float16* __restrict__ fbuf) {
  int k  = pi >> 6;          // 0..8
  int n  = pi & 63;
  int yl = n >> 5, xc = n & 31;
  int dy = k / 3,  dx = k % 3;
  int gy = y0 + yl + dy - 1;
  int gx = xc + dx - 1;
  float p = 0.0f;
  if (gy >= 0 && gy < HH && gx >= 0 && gx < WW)
    p = xch[gy * WW + gx];
  write_features(&fbuf[n * FS_STRIDE + k * NF], p);
}

// ---------------------------------------------------------------------------
// Fused KAN conv, software-pipelined: iteration i runs WMMA on feature buffer
// (i&1) while computing channel i+1's features into the other buffer; one
// barrier per channel, WMMA (XDL/TRANS pipe) co-executes with feature VALU.
// ---------------------------------------------------------------------------
__global__ __launch_bounds__(256)
void kan_conv_wmma(const float* __restrict__ x,
                   const _Float16* __restrict__ wf,
                   float* __restrict__ out) {
  __shared__ _Float16 fs[2][NTILE * FS_STRIDE];   // double-buffered [n][rp]

  const int tid  = threadIdx.x;
  const int lane = tid & 31;
  const int wv   = tid >> 5;          // wave 0..7
  const int mt   = wv & 3;            // M-tile (16 couts)
  const int nt0  = (wv >> 2) * 2;     // first of two N-tiles

  const int n0 = blockIdx.x * NTILE;
  const int b  = n0 >> 10;
  const int y0 = (n0 & 1023) >> 5;    // first of two output rows

  // zero the rp=90..95 pad rows in both buffers (never rewritten)
  if (tid < 192) {
    int n = tid / 3, d = tid % 3;
    *(uint32_t*)&fs[0][n * FS_STRIDE + 90 + d * 2] = 0u;
    *(uint32_t*)&fs[1][n * FS_STRIDE + 90 + d * 2] = 0u;
  }

  v8f acc0 = {};
  v8f acc1 = {};

  const int nlA = nt0 * 16 + (lane & 15);   // B column for tile nt0
  const int nlB = nlA + 16;                 // B column for tile nt0+1
  const int khi = (lane < 16) ? 0 : 8;      // B fragment K offset

  // prologue: features for channel 0 into buffer 0
  {
    const float* xch = x + (size_t)(b * CIN) * HH * WW;
    do_point(xch, y0, tid,       fs[0]);
    do_point(xch, y0, tid + 256, fs[0]);
    if (tid < 64) do_point(xch, y0, tid + 512, fs[0]);
  }
  __syncthreads();

  for (int i = 0; i < CIN; ++i) {
    _Float16* fcur = fs[i & 1];
    _Float16* fnxt = fs[(i & 1) ^ 1];

    // ---- 3 K-steps of 32 over this channel's 96 padded reduction rows
    const _Float16* wbase = wf + ((size_t)(i * 3 * 4 + mt) * 32 + lane) * 16;
    if (i + 1 < CIN)
      __builtin_prefetch(wbase + 3 * 4 * 32 * 16, 0, 1);  // next channel's A
    #pragma unroll
    for (int kc = 0; kc < 3; ++kc) {
      v16h a = *(const v16h*)(wbase + kc * (4 * 32 * 16));
      const _Float16* fb = fcur + kc * 32 + khi;
      v8h b0l = *(const v8h*)(fb + nlA * FS_STRIDE);
      v8h b0h = *(const v8h*)(fb + nlA * FS_STRIDE + 16);
      v8h b1l = *(const v8h*)(fb + nlB * FS_STRIDE);
      v8h b1h = *(const v8h*)(fb + nlB * FS_STRIDE + 16);
      v16h b0 = __builtin_shufflevector(b0l, b0h, 0,1,2,3,4,5,6,7,8,9,10,11,12,13,14,15);
      v16h b1 = __builtin_shufflevector(b1l, b1h, 0,1,2,3,4,5,6,7,8,9,10,11,12,13,14,15);
      acc0 = __builtin_amdgcn_wmma_f32_16x16x32_f16(false, a, false, b0,
                                                    (short)0, acc0, false, false);
      acc1 = __builtin_amdgcn_wmma_f32_16x16x32_f16(false, a, false, b1,
                                                    (short)0, acc1, false, false);
    }

    // ---- co-executing VALU: features for channel i+1 into the other buffer
    if (i + 1 < CIN) {
      const float* xch = x + (size_t)(b * CIN + (i + 1)) * HH * WW;
      do_point(xch, y0, tid,       fnxt);
      do_point(xch, y0, tid + 256, fnxt);
      if (tid < 64) do_point(xch, y0, tid + 512, fnxt);
    }
    __syncthreads();   // fnxt complete; all WMMA reads of fcur done
  }

  // ---- store D tiles per the 16x16 f32 C/D layout
  const int obase = mt * 16 + ((lane < 16) ? 0 : 8);
  #pragma unroll
  for (int tile = 0; tile < 2; ++tile) {
    v8f acc = tile ? acc1 : acc0;
    int nl = (nt0 + tile) * 16 + (lane & 15);
    int yy = y0 + (nl >> 5);
    int xc = nl & 31;
    #pragma unroll
    for (int v = 0; v < 8; ++v) {
      out[(((size_t)b * COUT + obase + v) * HH + yy) * WW + xc] = acc[v];
    }
  }
}

// ---------------------------------------------------------------------------
extern "C" void kernel_launch(void* const* d_in, const int* in_sizes, int n_in,
                              void* d_out, int out_size, void* d_ws, size_t ws_size,
                              hipStream_t stream) {
  const float* x = (const float*)d_in[0];   // (16,64,32,32)
  const float* w = (const float*)d_in[1];   // (64,64,9)
  const float* c = (const float*)d_in[2];   // (64,64,9,8)
  float* out = (float*)d_out;               // (16,64,32,32)
  _Float16* wf = (_Float16*)d_ws;           // 786432 bytes of swizzled f16 weights

  kan_prep_weights<<<(WF_THREADS + 255) / 256, 256, 0, stream>>>(w, c, wf);

  const int nblocks = (16 * HH * WW) / NTILE;   // 256 workgroups
  kan_conv_wmma<<<nblocks, 256, 0, stream>>>(x, wf, out);
}
